// FlowLayer_20220706030273
// MI455X (gfx1250) — matrix operand order repaired and checked by reading, not compiled, not run on hardware
//
#include <hip/hip_runtime.h>

typedef __attribute__((ext_vector_type(16))) __bf16 v16bf;
typedef __attribute__((ext_vector_type(8)))  __bf16 v8bf;
typedef __attribute__((ext_vector_type(8)))  float  v8f;

#define NB    8
#define CIN   256
#define COUT  64
#define HH    128
#define WW    128
#define HW    (HH*WW)          // 16384
#define NPIX  (NB*HW)          // 131072
#define ITERS 10

__device__ __forceinline__ float sigmoidf_(float v) {
    return 1.0f / (1.0f + __expf(-v));
}

// ---------------------------------------------------------------------------
// Kernel A: fused dual 1x1 conv (GEMM via bf16 WMMA) + sigmoid -> m1, m2 (NHWC)
// Per wave: one 16-pixel M-tile, N=64 for both m1 and m2 (8 accum tiles).
// B fragments are batch-loaded before the WMMA chain so the 8 WMMAs issue
// back-to-back without per-instruction dscnt stalls.
// ---------------------------------------------------------------------------
__global__ __launch_bounds__(256) void conv1_sigmoid_kernel(
    const float* __restrict__ x, const float* __restrict__ w1,
    const float* __restrict__ w2, float* __restrict__ m1,
    float* __restrict__ m2)
{
    // B staged fragment-major: [mat][k/16][n][k%16]; 2*16*64*16*2B = 64 KB LDS
    __shared__ __align__(32) __bf16 wlds[2][CIN/16][COUT][16];

    const int tid = threadIdx.x;
    for (int i = tid; i < CIN*COUT; i += 256) {
        int n = i / CIN;           // out channel
        int k = i % CIN;           // in channel
        wlds[0][k >> 4][n][k & 15] = (__bf16)w1[i];
        wlds[1][k >> 4][n][k & 15] = (__bf16)w2[i];
    }
    __syncthreads();

    const int wave = tid >> 5, lane = tid & 31;
    const int half = lane >> 4, lm = lane & 15;
    const int mt   = blockIdx.x * 8 + wave;      // M-tile (16 pixels)
    const int pix  = mt * 16 + lm;               // this lane's A row pixel
    const int nimg = pix >> 14;                  // / HW
    const int hw   = pix & (HW - 1);
    const float* xbase = x + (size_t)nimg * CIN * HW + hw;

    v8f acc[2][4] = {};

    #pragma unroll 2
    for (int ks = 0; ks < CIN / 32; ++ks) {
        // A fragment: contiguous k = 16*half + j (same k order as B from wlds)
        v16bf a;
        #pragma unroll
        for (int j = 0; j < 16; ++j) {
            int k = 16*half + j;
            a[j] = (__bf16)xbase[(size_t)(ks*32 + k) * HW];
        }
        // batch-load all 8 B fragments, then run the WMMA chain
        v16bf bfr[8];
        #pragma unroll
        for (int nt = 0; nt < 4; ++nt) {
            bfr[nt]     = *(const v16bf*)&wlds[0][ks*2 + half][nt*16 + lm][0];
            bfr[nt + 4] = *(const v16bf*)&wlds[1][ks*2 + half][nt*16 + lm][0];
        }
        #pragma unroll
        for (int nt = 0; nt < 4; ++nt) {
            acc[0][nt] = __builtin_amdgcn_wmma_f32_16x16x32_bf16(
                false, a, false, bfr[nt], (short)0, acc[0][nt], false, false);
            acc[1][nt] = __builtin_amdgcn_wmma_f32_16x16x32_bf16(
                false, a, false, bfr[nt + 4], (short)0, acc[1][nt], false, false);
        }
    }

    // C/D layout: VGPR r -> M = r + 8*half, N = lane%16. Store NHWC + sigmoid.
    const size_t pb = (size_t)mt * 16;
    #pragma unroll
    for (int nt = 0; nt < 4; ++nt) {
        #pragma unroll
        for (int r = 0; r < 8; ++r) {
            size_t p = pb + r + 8*half;
            int ch = nt*16 + lm;
            m1[p * 64 + ch] = sigmoidf_(acc[0][nt][r]);
            m2[p * 64 + ch] = sigmoidf_(acc[1][nt][r]);
        }
    }
}

// ---------------------------------------------------------------------------
// Kernel W: one grid_sample warp-update iteration for both tensors.
// One thread per (pixel, 4 channels): float4 gathers, weights computed once.
// ---------------------------------------------------------------------------
__device__ __forceinline__ float4 bilinear4_(const float4* __restrict__ mbase,
                                             float gx, float gy)
{
    float ix = ((gx + 1.0f) * (float)WW - 1.0f) * 0.5f;
    float iy = ((gy + 1.0f) * (float)HH - 1.0f) * 0.5f;
    float x0f = floorf(ix), y0f = floorf(iy);
    int x0 = (int)x0f, y0 = (int)y0f;
    float wx1 = ix - x0f, wy1 = iy - y0f;
    float wx0 = 1.0f - wx1, wy0 = 1.0f - wy1;
    float4 acc = make_float4(0.f, 0.f, 0.f, 0.f);
    #pragma unroll
    for (int dy = 0; dy < 2; ++dy) {
        int yi = y0 + dy;
        if (yi < 0 || yi > HH - 1) continue;
        float wy = dy ? wy1 : wy0;
        #pragma unroll
        for (int dx = 0; dx < 2; ++dx) {
            int xi = x0 + dx;
            if (xi < 0 || xi > WW - 1) continue;
            float wv = wy * (dx ? wx1 : wx0);
            float4 s = mbase[(size_t)(yi * WW + xi) * 16];   // stride 64 floats
            acc.x += wv * s.x;  acc.y += wv * s.y;
            acc.z += wv * s.z;  acc.w += wv * s.w;
        }
    }
    return acc;
}

__global__ __launch_bounds__(256) void warp_kernel(
    const float* __restrict__ grads,
    const float* __restrict__ s1, const float* __restrict__ s2,
    float* __restrict__ d1, float* __restrict__ d2)
{
    int idx = blockIdx.x * 256 + threadIdx.x;   // over NPIX*16
    int c4 = (idx & 15) * 4;                    // channel group
    int p  = idx >> 4;
    int n  = p >> 14;
    int hw = p & (HW - 1);

    float gx = grads[((size_t)n*2 + 0) * HW + hw] * (1.0f / (float)ITERS);
    float gy = grads[((size_t)n*2 + 1) * HW + hw] * (1.0f / (float)ITERS);

    const float4* b1 = (const float4*)(s1 + (size_t)n * HW * 64 + c4);
    const float4* b2 = (const float4*)(s2 + (size_t)n * HW * 64 + c4);

    float4 i1 = *(const float4*)(s1 + (size_t)p * 64 + c4);
    float4 i2 = *(const float4*)(s2 + (size_t)p * 64 + c4);
    float4 a1 = bilinear4_(b1,  gx,  gy);
    float4 a2 = bilinear4_(b2, -gx, -gy);

    float4 o1 = make_float4(i1.x + a1.x, i1.y + a1.y, i1.z + a1.z, i1.w + a1.w);
    float4 o2 = make_float4(i2.x + a2.x, i2.y + a2.y, i2.z + a2.z, i2.w + a2.w);
    *(float4*)(d1 + (size_t)p * 64 + c4) = o1;
    *(float4*)(d2 + (size_t)p * 64 + c4) = o2;
}

// ---------------------------------------------------------------------------
// Kernel C: fused sigmoid + depthwise 3x3 + pointwise 1x1.
// Phase 1: each wave computes the depthwise output for its 16-pixel tile
//   (128 channels), staged into LDS as bf16 in WMMA A-fragment layout
//   [ktile][pix][k%16] -> each lane's A fragment is two 16B ds_load_b128.
// Phase 2: WMMA GEMM [16 x 128] x [128 x 64] -> y (NCHW).
// ---------------------------------------------------------------------------
__global__ __launch_bounds__(256) void dwpw_kernel(
    const float* __restrict__ m1, const float* __restrict__ m2,
    const float* __restrict__ dwt, const float* __restrict__ pwt,
    float* __restrict__ y)
{
    __shared__ __align__(32) __bf16 plds[8][COUT][16];     // pw weights, 16 KB
    __shared__ __align__(32) __bf16 alds[8][8][16][16];    // A tiles, 32 KB

    const int tid = threadIdx.x;
    for (int i = tid; i < COUT * 128; i += 256) {
        int n = i >> 7, k = i & 127;
        plds[k >> 4][n][k & 15] = (__bf16)pwt[i];
    }

    const int wave = tid >> 5, lane = tid & 31;
    const int mt   = blockIdx.x * 8 + wave;
    const size_t pb = (size_t)mt * 16;

    // ---- phase 1: depthwise (2 channels per lane per pass) ----
    const int cb = 2 * lane;               // 0..62 within the 64-ch tensor
    #pragma unroll 1
    for (int pix = 0; pix < 16; ++pix) {
        size_t p = pb + pix;
        int n  = (int)(p >> 14);
        int hw = (int)(p & (HW - 1));
        int h  = hw >> 7, w = hw & 127;
        #pragma unroll
        for (int t2 = 0; t2 < 2; ++t2) {   // 0: m1 (ch 0..63), 1: m2 (64..127)
            const float* m = t2 ? m2 : m1;
            const float2* base =
                (const float2*)(m + (size_t)n * HW * 64 + cb);
            int c0 = t2 * 64 + cb;
            float a0 = 0.f, a1 = 0.f;
            #pragma unroll
            for (int dy = -1; dy <= 1; ++dy) {
                int hhp = h + dy;
                if (hhp < 0 || hhp >= HH) continue;
                #pragma unroll
                for (int dx = -1; dx <= 1; ++dx) {
                    int wi = w + dx;
                    if (wi < 0 || wi >= WW) continue;
                    int j = (dy + 1) * 3 + (dx + 1);
                    float2 v = base[(size_t)(hhp * WW + wi) * 32];
                    a0 += dwt[(c0    ) * 9 + j] * sigmoidf_(v.x);
                    a1 += dwt[(c0 + 1) * 9 + j] * sigmoidf_(v.y);
                }
            }
            alds[wave][c0 >> 4][pix][(c0 & 15)    ] = (__bf16)a0;
            alds[wave][c0 >> 4][pix][(c0 & 15) + 1] = (__bf16)a1;
        }
    }
    __syncthreads();

    // ---- phase 2: pointwise GEMM via WMMA ----
    const int half = lane >> 4, lm = lane & 15;
    v8f acc[4] = {};
    #pragma unroll
    for (int ks = 0; ks < 4; ++ks) {
        // lane A fragment: K = ks*32 + 16*half + 0..15
        v8bf lo = *(const v8bf*)&alds[wave][ks*2 + half][lm][0];
        v8bf hi = *(const v8bf*)&alds[wave][ks*2 + half][lm][8];
        v16bf a;
        #pragma unroll
        for (int j = 0; j < 8; ++j) { a[j] = lo[j]; a[j + 8] = hi[j]; }

        v16bf bfr[4];
        #pragma unroll
        for (int nt = 0; nt < 4; ++nt)
            bfr[nt] = *(const v16bf*)&plds[ks*2 + half][nt*16 + lm][0];
        #pragma unroll
        for (int nt = 0; nt < 4; ++nt)
            acc[nt] = __builtin_amdgcn_wmma_f32_16x16x32_bf16(
                false, a, false, bfr[nt], (short)0, acc[nt], false, false);
    }

    #pragma unroll
    for (int nt = 0; nt < 4; ++nt) {
        #pragma unroll
        for (int r = 0; r < 8; ++r) {
            size_t p = pb + r + 8*half;
            int ch = nt*16 + lm;
            int n  = (int)(p >> 14);
            int hw = (int)(p & (HW - 1));
            y[((size_t)n * COUT + ch) * HW + hw] = acc[nt][r];
        }
    }
}

// ---------------------------------------------------------------------------
extern "C" void kernel_launch(void* const* d_in, const int* in_sizes, int n_in,
                              void* d_out, int out_size, void* d_ws, size_t ws_size,
                              hipStream_t stream)
{
    const float* x     = (const float*)d_in[0];
    const float* grads = (const float*)d_in[1];
    const float* w1    = (const float*)d_in[2];
    const float* w2    = (const float*)d_in[3];
    const float* dwt   = (const float*)d_in[4];
    const float* pwt   = (const float*)d_in[5];
    float* y = (float*)d_out;

    const size_t msz = (size_t)NPIX * 64;   // floats per m tensor
    float* m1a = (float*)d_ws;
    float* m2a = m1a + msz;
    float* m1b = m2a + msz;
    float* m2b = m1b + msz;

    // Stage 1: fused dual 1x1 conv + sigmoid (bf16 WMMA GEMM)
    conv1_sigmoid_kernel<<<8192 / 8, 256, 0, stream>>>(x, w1, w2, m1a, m2a);

    // Stage 2: 10 serialized grid_sample updates (ping-pong, L2-resident)
    for (int it = 0; it < ITERS; ++it) {
        const float* s1 = (it & 1) ? m1b : m1a;
        const float* s2 = (it & 1) ? m2b : m2a;
        float* dd1      = (it & 1) ? m1a : m1b;
        float* dd2      = (it & 1) ? m2a : m2b;
        warp_kernel<<<(NPIX * 16) / 256, 256, 0, stream>>>(grads, s1, s2, dd1, dd2);
    }

    // Stage 3: fused sigmoid + depthwise 3x3 + pointwise WMMA GEMM -> y (NCHW)
    dwpw_kernel<<<8192 / 8, 256, 0, stream>>>(m1a, m2a, dwt, pwt, y);
}
